// LightGCN_ecc_6021544149550
// MI455X (gfx1250) — matrix operand order repaired and checked by reading, not compiled, not run on hardware
//
#include <hip/hip_runtime.h>
#include <hip/hip_bf16.h>

// ---------------------------------------------------------------------------
// LightGCN on MI455X (gfx1250, wave32, WMMA).
//
//   1. L0 = concat(user_emb, item_emb); L1..L3 zeroed
//   2. 3x SpMM: one wave32 per edge, float2 gather + atomicAdd scatter (L2-resident)
//   3. einsum+W collapsed into A(256x256) = sum_i W[i,j]*u_layer_i, pre-packed
//      into the WMMA 16-bit A-operand lane layout (fp16)
//   4. V(50000x256) fp16 (k = layer*64 + dim)
//   5. GEMM 256x50000x256: B tile staged to LDS via global_load_async_to_lds
//      (ASYNCcnt path), consumed by v_wmma_f32_16x16x32_f16, fused bias+sigmoid
//      (v_exp_f32 + v_rcp_f32 fast epilogue)
// ---------------------------------------------------------------------------

#define N_USERS   60000
#define M_ITEMS   50000
#define EMBED_DIM 64
#define N_LAYERS  3
#define NNZ_CNT   3500000
#define B_USERS   256
#define N_TOTAL   (N_USERS + M_ITEMS)
#define KDIM      256                              // (N_LAYERS+1) * EMBED_DIM
#define LELEMS    ((size_t)N_TOTAL * EMBED_DIM)    // floats per layer buffer
#define NGROUPS   782                              // ceil(50000 / 64)
#define LDS_ROW_H 264                              // 528 B row stride (16B pad): banks 4*nl

#define USE_ASYNC_LDS 1

typedef __attribute__((ext_vector_type(16))) _Float16 v16h;
typedef __attribute__((ext_vector_type(8)))  _Float16 v8h;
typedef __attribute__((ext_vector_type(8)))  float    v8f;

// ---------------------------------------------------------------------------
// Phase 1: init layer 0, zero layers 1..3
// ---------------------------------------------------------------------------
__global__ void lgcn_init_l0(const float* __restrict__ ue,
                             const float* __restrict__ ie,
                             float* __restrict__ L0) {
    size_t idx = (size_t)blockIdx.x * blockDim.x + threadIdx.x;
    if (idx >= LELEMS) return;
    const size_t usplit = (size_t)N_USERS * EMBED_DIM;
    L0[idx] = (idx < usplit) ? ue[idx] : ie[idx - usplit];
}

__global__ void lgcn_zero(float* __restrict__ p, size_t n) {
    size_t idx = (size_t)blockIdx.x * blockDim.x + threadIdx.x;
    if (idx < n) p[idx] = 0.0f;
}

// ---------------------------------------------------------------------------
// Phase 2: SpMM scatter.  One wave32 per edge; lane handles 2 dims (float2).
// Wave reads 256 contiguous bytes of the source row; atomics land in L2
// (28 MB embedding table << 192 MB L2).
// ---------------------------------------------------------------------------
__global__ __launch_bounds__(256)
void lgcn_spmm(const int*   __restrict__ row,
               const int*   __restrict__ col,
               const float* __restrict__ vals,
               const float* __restrict__ Lin,
               float*       __restrict__ Lout) {
    size_t gid  = (size_t)blockIdx.x * blockDim.x + threadIdx.x;
    size_t e    = gid >> 5;
    int    lane = (int)(gid & 31);
    if (e >= NNZ_CNT) return;
    int   src = col[e];
    int   dst = row[e];
    float v   = vals[e];
    const float2 s = *(const float2*)(Lin + (size_t)src * EMBED_DIM + lane * 2);
    float* dp = Lout + (size_t)dst * EMBED_DIM + lane * 2;
    atomicAdd(dp + 0, v * s.x);
    atomicAdd(dp + 1, v * s.y);
}

// ---------------------------------------------------------------------------
// Phase 3: A[b, j*64+d] = sum_i W[i*4+j] * L_i[users[b]*64 + d], written in
// the 16-bit A-operand layout (ISA 7.12.2, 16x32 fragment):
//   lane = g*16 + m ; half t = 2v+h with
//   v = ((kin>>2)&4)|((kin>>1)&3), g = (kin>>3)&1, h = kin&1
// so the GEMM loads each fragment as one contiguous v16h per lane.
// ---------------------------------------------------------------------------
__global__ void lgcn_build_a(const int*   __restrict__ users,
                             const float* __restrict__ W,
                             const float* __restrict__ Lbase,
                             _Float16*    __restrict__ Apk) {
    int idx = blockIdx.x * blockDim.x + threadIdx.x;   // 65536 = 256*256
    int b   = idx >> 8;
    int k   = idx & 255;
    int j   = k >> 6;
    int d   = k & 63;
    int u   = users[b];
    float a = 0.0f;
#pragma unroll
    for (int i = 0; i <= N_LAYERS; ++i)
        a += W[i * (N_LAYERS + 1) + j] *
             Lbase[(size_t)i * LELEMS + (size_t)u * EMBED_DIM + d];
    int mt    = b >> 4, m = b & 15;
    int chunk = k >> 5, kin = k & 31;
    int g = (kin >> 3) & 1;
    int v = ((kin >> 2) & 4) | ((kin >> 1) & 3);
    int h = kin & 1;
    int lane = g * 16 + m;
    size_t pos = ((size_t)((mt * 8 + chunk) * 32 + lane) << 4) + (size_t)(2 * v + h);
    Apk[pos] = (_Float16)a;
}

// ---------------------------------------------------------------------------
// Phase 4: V[m, j*64+d] = L_j[(N_USERS+m)*64 + d] in fp16 (row-major, k contig)
// ---------------------------------------------------------------------------
__global__ void lgcn_conv_v(const float* __restrict__ Lbase,
                            _Float16*    __restrict__ Vh) {
    size_t idx = (size_t)blockIdx.x * blockDim.x + threadIdx.x;
    if (idx >= (size_t)M_ITEMS * KDIM) return;
    size_t m = idx >> 8;
    int    k = (int)(idx & 255);
    int    j = k >> 6;
    int    d = k & 63;
    Vh[idx] = (_Float16)Lbase[(size_t)j * LELEMS +
                              ((size_t)N_USERS + m) * EMBED_DIM + d];
}

// ---------------------------------------------------------------------------
// Phase 5: GEMM  out(256 x 50000) = sigmoid(A(256x256) @ V^T + bias)
// Block = 8 waves; block (ng, half) stages the 64-col B tile into LDS once
// (async global->LDS), every wave then runs one 16-row M-tile over 4 N-tiles.
// Loads are unconditional (tail columns clamped); only stores are predicated,
// so WMMA always executes with full EXEC.
// ---------------------------------------------------------------------------
__global__ __launch_bounds__(256)
void lgcn_gemm(const _Float16* __restrict__ Apk,
               const _Float16* __restrict__ Vh,
               const float*    __restrict__ bptr,
               float*          __restrict__ out) {
    __shared__ _Float16 Btile[64 * LDS_ROW_H];      // 33792 B, padded rows

    const int tid  = (int)threadIdx.x;
    const int lane = tid & 31;
    const int wave = tid >> 5;
    const int ng   = (int)blockIdx.x >> 1;          // N-group of 64 cols
    const int mt   = ((int)blockIdx.x & 1) * 8 + wave;   // M-tile 0..15
    const float bias = bptr[0];

    // ---- stage B tile: 64 rows x 512B -> LDS rows of 528B ------------------
    // 2048 16-byte units; each of 256 threads copies 8 units.
#pragma unroll
    for (int w = 0; w < 8; ++w) {
        int unit = tid + w * 256;
        int r = unit >> 5;                 // tile row (column of V)
        int u = unit & 31;                 // 16B unit within row
        int colc = ng * 64 + r;
        if (colc > M_ITEMS - 1) colc = M_ITEMS - 1;   // clamp tail (dup data)
        const void* gsrc = (const void*)(Vh + (size_t)colc * KDIM + u * 8);
        unsigned ldst = (unsigned)(size_t)(void*)Btile + (unsigned)(r * 528 + u * 16);
#if USE_ASYNC_LDS
        asm volatile("global_load_async_to_lds_b128 %0, %1, off"
                     :: "v"(ldst), "v"(gsrc) : "memory");
#else
        *(float4*)((char*)nullptr + ldst) = *(const float4*)gsrc;  // unused path
#endif
    }
#if USE_ASYNC_LDS
    asm volatile("s_wait_asynccnt 0x0" ::: "memory");
#endif
    __syncthreads();

    const int g  = lane >> 4;              // K-half / M-half selector
    const int nl = lane & 15;

    // ---- A fragments: contiguous per-lane v16h loads, kept in VGPRs --------
    v16h afrag[8];
#pragma unroll
    for (int c = 0; c < 8; ++c)
        afrag[c] = *(const v16h*)(Apk + (((size_t)(mt * 8 + c) * 32 + lane) << 4));

#pragma unroll
    for (int nt = 0; nt < 4; ++nt) {
        const int colv = ng * 64 + nt * 16 + nl;
        const _Float16* brow = Btile + (nt * 16 + nl) * LDS_ROW_H + g * 8;

        // preload all 8 B fragments (two 16B LDS loads each; rows are 16B-aligned)
        union Frag { v16h v; v8h h[2]; };
        Frag bfrag[8];
#pragma unroll
        for (int c = 0; c < 8; ++c) {
            bfrag[c].h[0] = *(const v8h*)(brow + c * 32);
            bfrag[c].h[1] = *(const v8h*)(brow + c * 32 + 16);
        }

        v8f acc = {};
#pragma unroll
        for (int c = 0; c < 8; ++c)
            acc = __builtin_amdgcn_wmma_f32_16x16x32_f16(
                      /*neg_a=*/false, afrag[c],
                      /*neg_b=*/false, bfrag[c].v,
                      /*c_mod=*/(short)0, acc,
                      /*reuse_a=*/false, /*reuse_b=*/false);

        if (colv < M_ITEMS) {              // predicate stores only
#pragma unroll
            for (int r = 0; r < 8; ++r) {
                const int rowo = mt * 16 + g * 8 + r;   // C layout: VGPR r -> M = r + 8*g
                const float x = acc[r] + bias;
                // fast sigmoid: v_exp_f32 + v_rcp_f32 (no IEEE divide expansion)
                out[(size_t)rowo * M_ITEMS + colv] =
                    __builtin_amdgcn_rcpf(1.0f + __expf(-x));
            }
        }
    }
}

// ---------------------------------------------------------------------------
// Launch
// ---------------------------------------------------------------------------
extern "C" void kernel_launch(void* const* d_in, const int* in_sizes, int n_in,
                              void* d_out, int out_size, void* d_ws, size_t ws_size,
                              hipStream_t stream) {
    (void)in_sizes; (void)n_in; (void)out_size; (void)ws_size;

    const int*   row      = (const int*)  d_in[0];
    const int*   col      = (const int*)  d_in[1];
    const float* vals     = (const float*)d_in[2];
    const int*   users    = (const int*)  d_in[3];
    const float* user_emb = (const float*)d_in[4];
    const float* item_emb = (const float*)d_in[5];
    const float* W        = (const float*)d_in[6];
    const float* bias     = (const float*)d_in[7];
    float*       out      = (float*)d_out;

    // workspace layout
    char* ws = (char*)d_ws;
    float*    L   = (float*)ws;                                   // 4 * LELEMS f32 (112.6 MB)
    _Float16* Apk = (_Float16*)(ws + 4 * LELEMS * sizeof(float)); // 65536 f16
    _Float16* Vh  = Apk + (size_t)B_USERS * KDIM;                 // 50000*256 f16 (25.6 MB)

    // 1. init L0, zero L1..L3
    {
        int blk = (int)((LELEMS + 255) / 256);
        lgcn_init_l0<<<blk, 256, 0, stream>>>(user_emb, item_emb, L);
        size_t nz = 3 * LELEMS;
        lgcn_zero<<<(int)((nz + 255) / 256), 256, 0, stream>>>(L + LELEMS, nz);
    }

    // 2. three propagation layers (wave per edge)
    {
        size_t threads = (size_t)NNZ_CNT * 32;
        int blk = (int)((threads + 255) / 256);
        for (int l = 1; l <= N_LAYERS; ++l)
            lgcn_spmm<<<blk, 256, 0, stream>>>(row, col, vals,
                                               L + (size_t)(l - 1) * LELEMS,
                                               L + (size_t)l * LELEMS);
    }

    // 3. fold W into the user side -> packed WMMA A operand
    lgcn_build_a<<<(B_USERS * KDIM) / 256, 256, 0, stream>>>(users, W, L, Apk);

    // 4. fp16 item matrix
    {
        size_t n = (size_t)M_ITEMS * KDIM;
        lgcn_conv_v<<<(int)((n + 255) / 256), 256, 0, stream>>>(L, Vh);
    }

    // 5. WMMA GEMM + sigmoid.  782 N-groups x 2 half-blocks = 1564 blocks.
    lgcn_gemm<<<NGROUPS * 2, 256, 0, stream>>>(Apk, Vh, bias, out);
}
